// MultiheadedLinearSelfAttention_58256936403003
// MI455X (gfx1250) — matrix operand order repaired and checked by reading, not compiled, not run on hardware
//
#include <hip/hip_runtime.h>
#include <hip/hip_bf16.h>

// MI455X / gfx1250, wave32. Matrix ops via v_wmma_f32_16x16x32_bf16,
// tile staging via GLOBAL_LOAD_ASYNC_TO_LDS_B128 (ASYNCcnt) when available.

typedef __attribute__((ext_vector_type(16))) __bf16 v16bf;
typedef __attribute__((ext_vector_type(8)))  __bf16 v8bf;
typedef __attribute__((ext_vector_type(8)))  float  v8f;
typedef int gv4i __attribute__((vector_size(16)));   // matches builtin pointee

#define Bc    4
#define Sc    4096
#define Ec    1024
#define Hc    16
#define Dc    64
#define TOKc  (Bc * Sc)      // 16384
#define E3c   (3 * Ec)       // 3072
#define SPLIT 8              // K2 split-K factor over S

union FragU { v16bf v; v8bf h[2]; };

__device__ __forceinline__ v16bf pack2(v8bf lo, v8bf hi) {
    FragU u; u.h[0] = lo; u.h[1] = hi; return u.v;
}

__device__ __forceinline__ float phi_f(float x) {
    // elu(x)+1 == x+1 (x>0) else exp(x)
    return x > 0.0f ? x + 1.0f : __expf(x);
}

// ---- async global->LDS copy (16B per lane), ASYNCcnt-tracked ------------
#if defined(__gfx1250__) && __has_builtin(__builtin_amdgcn_global_load_async_to_lds_b128)
#define HAS_ASYNC_LDS 1
#else
#define HAS_ASYNC_LDS 0
#endif

__device__ __forceinline__ void async_copy_b128(const __bf16* gsrc, __bf16* ldst) {
#if HAS_ASYNC_LDS
    __builtin_amdgcn_global_load_async_to_lds_b128(
        (__attribute__((address_space(1))) gv4i*)(uintptr_t)gsrc,
        (__attribute__((address_space(3))) gv4i*)(uintptr_t)(uint32_t)(uintptr_t)ldst,
        0, 0);
#else
    *(v8bf*)ldst = *(const v8bf*)gsrc;
#endif
}

__device__ __forceinline__ void async_wait_all() {
#if HAS_ASYNC_LDS
#if __has_builtin(__builtin_amdgcn_s_wait_asynccnt)
    __builtin_amdgcn_s_wait_asynccnt(0);
#else
    asm volatile("s_wait_asynccnt 0x0" ::: "memory");
#endif
#endif
}

// ---------------------------------------------------------------- K0: cast
__global__ void f32_to_bf16_v8(const float* __restrict__ in,
                               __bf16* __restrict__ out, int n8) {
    int i = blockIdx.x * blockDim.x + threadIdx.x;
    int stride = gridDim.x * blockDim.x;
    for (; i < n8; i += stride) {
        const float4* s = (const float4*)in + (size_t)i * 2;
        float4 a = s[0], b = s[1];
        v8bf o;
        o[0] = (__bf16)a.x; o[1] = (__bf16)a.y; o[2] = (__bf16)a.z; o[3] = (__bf16)a.w;
        o[4] = (__bf16)b.x; o[5] = (__bf16)b.y; o[6] = (__bf16)b.z; o[7] = (__bf16)b.w;
        ((v8bf*)out)[i] = o;
    }
}

// ---------------------------------------------------------------- K1: qkv GEMM
// C(16384 x 3072) = X(16384 x 1024) * W(1024 x 3072), double-buffered LDS,
// A-panel staged via async-to-LDS, B-panel prefetched in registers across
// the WMMA block. Epilogue: bias + phi, scatter bf16 into pq / pk / v.
#define LDA 40   // LDS row stride (elems); 80 B rows, 16B-aligned
__global__ __launch_bounds__(256) void qkv_gemm_kernel(
    const __bf16* __restrict__ X, const __bf16* __restrict__ W,
    const float* __restrict__ bq,
    __bf16* __restrict__ PQ, __bf16* __restrict__ PK, __bf16* __restrict__ V) {

    __shared__ __align__(16) __bf16 As[2][128 * LDA];  // [row m][k]
    __shared__ __align__(16) __bf16 Bs[2][128 * LDA];  // [col n][k] (transposed)

    const int t    = threadIdx.x;
    const int lane = t & 31;
    const int wave = t >> 5;
    const int m0 = blockIdx.y * 128;
    const int n0 = blockIdx.x * 128;
    const int wm = (wave & 3) * 32;   // wave's 32-row strip
    const int wn = (wave >> 2) * 64;  // wave's 64-col strip

    v8f acc[2][4];
    #pragma unroll
    for (int i = 0; i < 2; i++)
        #pragma unroll
        for (int j = 0; j < 4; j++) acc[i][j] = {};

    const int ar = t >> 1, ah = (t & 1) * 16;   // A stager: 2x16B / thread
    const int bk = t >> 3, bs = (t & 7) * 16;   // B stager: 16 elems / thread

    // ---- prologue: stage tile kt=0 into buffer 0
    {
        const __bf16* srcA = X + (size_t)(m0 + ar) * Ec + ah;
        async_copy_b128(srcA,     &As[0][ar * LDA + ah]);
        async_copy_b128(srcA + 8, &As[0][ar * LDA + ah + 8]);
        const __bf16* srcB = W + (size_t)bk * E3c + n0 + bs;
        v8bf b0 = *(const v8bf*)(srcB);
        v8bf b1 = *(const v8bf*)(srcB + 8);
        #pragma unroll
        for (int e = 0; e < 8; e++) Bs[0][(bs + e) * LDA + bk]     = b0[e];
        #pragma unroll
        for (int e = 0; e < 8; e++) Bs[0][(bs + 8 + e) * LDA + bk] = b1[e];
        async_wait_all();
        __syncthreads();
    }

    int cur = 0;
    for (int kt = 0; kt < Ec; kt += 32) {
        const bool pre = (kt + 32) < Ec;
        const int nxt = cur ^ 1;
        v8bf nb0 = {}, nb1 = {};
        if (pre) {
            // issue next B-panel global loads (consumed after the WMMAs)
            const __bf16* srcB = W + (size_t)(kt + 32 + bk) * E3c + n0 + bs;
            nb0 = *(const v8bf*)(srcB);
            nb1 = *(const v8bf*)(srcB + 8);
            // next A-panel: async DMA straight into the other LDS buffer
            const __bf16* srcA = X + (size_t)(m0 + ar) * Ec + kt + 32 + ah;
            async_copy_b128(srcA,     &As[nxt][ar * LDA + ah]);
            async_copy_b128(srcA + 8, &As[nxt][ar * LDA + ah + 8]);
        }

        // Pack fragments with A and B both contiguous-k (lane<16: K0..15;
        // lane>=16: K16..31).
        v16bf fa[2], fb[4];
        #pragma unroll
        for (int i = 0; i < 2; i++) {
            int r  = wm + i * 16 + (lane & 15);
            int kk = (lane >> 4) * 16;
            fa[i] = pack2(*(const v8bf*)&As[cur][r * LDA + kk],
                          *(const v8bf*)&As[cur][r * LDA + kk + 8]);
        }
        #pragma unroll
        for (int j = 0; j < 4; j++) {
            int c  = wn + j * 16 + (lane & 15);
            int kk = (lane >> 4) * 16;
            fb[j] = pack2(*(const v8bf*)&Bs[cur][c * LDA + kk],
                          *(const v8bf*)&Bs[cur][c * LDA + kk + 8]);
        }
        #pragma unroll
        for (int i = 0; i < 2; i++)
            #pragma unroll
            for (int j = 0; j < 4; j++)
                acc[i][j] = __builtin_amdgcn_wmma_f32_16x16x32_bf16(
                    false, fa[i], false, fb[j], (short)0, acc[i][j], false, false);

        if (pre) {   // transpose-scatter the prefetched B-panel
            #pragma unroll
            for (int e = 0; e < 8; e++) Bs[nxt][(bs + e) * LDA + bk]     = nb0[e];
            #pragma unroll
            for (int e = 0; e < 8; e++) Bs[nxt][(bs + 8 + e) * LDA + bk] = nb1[e];
        }
        async_wait_all();
        __syncthreads();
        cur ^= 1;
    }

    // Epilogue: C layout = VGPR r -> row r (+8 for lanes>=16), col = lane&15.
    // part is uniform per 16-col tile -> scalar branch, no per-lane selects.
    #pragma unroll
    for (int i = 0; i < 2; i++) {
        #pragma unroll
        for (int j = 0; j < 4; j++) {
            const int colbase = n0 + wn + j * 16;
            const int part = colbase >> 10;              // 0:q 1:k 2:v
            const int rem  = (colbase & 1023) + (lane & 15);
            __bf16* dst = (part == 0) ? PQ : (part == 1) ? PK : V;
            const float bias = bq[colbase + (lane & 15)];
            const int row0 = m0 + wm + i * 16 + ((lane >> 4) * 8);
            if (part < 2) {
                #pragma unroll
                for (int r = 0; r < 8; r++)
                    dst[(size_t)(row0 + r) * Ec + rem] =
                        (__bf16)phi_f(acc[i][j][r] + bias);
            } else {
                #pragma unroll
                for (int r = 0; r < 8; r++)
                    dst[(size_t)(row0 + r) * Ec + rem] =
                        (__bf16)(acc[i][j][r] + bias);
            }
        }
    }
}

// ---------------------------------------------------------------- K2: kv + ksum (split-K over S)
// Per (b,h,chunk): partial kvT[d][m] = sum_s pk[s][d]*v[s][m] over 512 rows,
// plus partial ksum[d]. 512 WGs so the 67MB stream can saturate HBM.
#define LDP 40
__global__ __launch_bounds__(256) void kv_ksum_kernel(
    const __bf16* __restrict__ PK, const __bf16* __restrict__ V,
    float* __restrict__ KVP, float* __restrict__ KSP) {

    __shared__ __align__(16) __bf16 pks[64 * LDP];  // [d][s_chunk]
    __shared__ __align__(16) __bf16 vs [64 * LDP];  // [m][s_chunk]

    const int t = threadIdx.x, lane = t & 31, wave = t >> 5;
    const int bh = blockIdx.x / SPLIT, chunk = blockIdx.x % SPLIT;
    const int b = bh >> 4, h = bh & 15;
    const int sbeg = chunk * (Sc / SPLIT);
    const int i  = (wave & 3) * 16;   // d-tile base
    const int j0 = (wave >> 2) * 2;   // two m-tiles

    v8f acc[2]; acc[0] = {}; acc[1] = {};
    float ks = 0.f;

    const int r = t >> 3, seg = (t & 7) * 8;

    for (int sc = sbeg; sc < sbeg + Sc / SPLIT; sc += 32) {
        size_t gro = (size_t)(b * Sc + sc + r) * Ec + h * Dc + seg;
        v8bf p = *(const v8bf*)(PK + gro);
        v8bf q = *(const v8bf*)(V  + gro);
        #pragma unroll
        for (int e = 0; e < 8; e++) {
            pks[(seg + e) * LDP + r] = p[e];
            vs [(seg + e) * LDP + r] = q[e];
        }
        __syncthreads();

        if (t < 64) {
            #pragma unroll
            for (int rr = 0; rr < 32; rr++) ks += (float)pks[t * LDP + rr];
        }

        int kk8 = (lane >> 4) * 16;
        int da  = i + (lane & 15);
        v16bf fa = pack2(*(const v8bf*)&pks[da * LDP + kk8],
                         *(const v8bf*)&pks[da * LDP + kk8 + 8]);
        int kk16 = (lane >> 4) * 16;
        #pragma unroll
        for (int jj = 0; jj < 2; jj++) {
            int m = (j0 + jj) * 16 + (lane & 15);
            v16bf fbv = pack2(*(const v8bf*)&vs[m * LDP + kk16],
                              *(const v8bf*)&vs[m * LDP + kk16 + 8]);
            acc[jj] = __builtin_amdgcn_wmma_f32_16x16x32_bf16(
                false, fa, false, fbv, (short)0, acc[jj], false, false);
        }
        __syncthreads();
    }

    const size_t kvstride = (size_t)Bc * Hc * Dc * Dc;   // 262144
    #pragma unroll
    for (int jj = 0; jj < 2; jj++) {
        int m = (j0 + jj) * 16 + (lane & 15);
        #pragma unroll
        for (int rr = 0; rr < 8; rr++) {
            int d = i + rr + ((lane >> 4) * 8);
            KVP[chunk * kvstride + ((size_t)bh * Dc + d) * Dc + m] = acc[jj][rr];
        }
    }
    if (t < 64) KSP[(chunk * Bc * Hc + bh) * Dc + t] = ks;
}

// ---------------------------------------------------------------- K2b: deterministic split-K reduce
__global__ void kv_reduce_kernel(const float* __restrict__ KVP,
                                 const float* __restrict__ KSP,
                                 float* __restrict__ KVF,
                                 float* __restrict__ KSUM,
                                 int nkv, int nks) {
    int idx = blockIdx.x * blockDim.x + threadIdx.x;
    if (idx < nkv) {
        float s = 0.f;
        #pragma unroll
        for (int c = 0; c < SPLIT; c++) s += KVP[(size_t)c * nkv + idx];
        KVF[idx] = s;
    }
    if (idx < nks) {
        float s = 0.f;
        #pragma unroll
        for (int c = 0; c < SPLIT; c++) s += KSP[(size_t)c * nks + idx];
        KSUM[idx] = s;
    }
}

// ---------------------------------------------------------------- K3: attn
// Per (b,h,128 rows): stage pq tile once (async), z = 1/(pq.ksum+1e-6) from
// LDS, scale tile in place, attn = (z*pq)(128x64) @ kvT(64x64) via WMMA.
#define LDQ 72
__global__ __launch_bounds__(256) void attn_kernel(
    const __bf16* __restrict__ PQ, const float* __restrict__ KVF,
    const float* __restrict__ KSUM, float* __restrict__ OUT) {

    __shared__ __align__(16) __bf16 zpq[128 * LDQ];  // [s][d]
    __shared__ __align__(16) __bf16 kvt[64 * LDQ];   // [m][d] (transposed)
    __shared__ float ksl[64];
    __shared__ float zrow[128];

    const int t = threadIdx.x, lane = t & 31, wave = t >> 5;
    const int bh = blockIdx.x >> 5, sblk = blockIdx.x & 31;
    const int b = bh >> 4, h = bh & 15;
    const int s0 = sblk * 128;

    // stage pq tile: async 4x16B per thread, [s][d] layout
    const int rr = t >> 1, d0 = (t & 1) * 32;
    {
        const __bf16* prow = PQ + (size_t)(b * Sc + s0 + rr) * Ec + h * Dc + d0;
        #pragma unroll
        for (int c = 0; c < 4; c++)
            async_copy_b128(prow + c * 8, &zpq[rr * LDQ + d0 + c * 8]);
    }
    if (t < 64) ksl[t] = KSUM[bh * Dc + t];
    {   // stage kvT(64x64) f32 -> bf16, transposed -> kvt[m][d]
        int d = t >> 2, seg = (t & 3) * 16;
        const float4* src = (const float4*)&KVF[((size_t)bh * Dc + d) * Dc + seg];
        #pragma unroll
        for (int c = 0; c < 4; c++) {
            float4 v4 = src[c];
            kvt[(seg + c * 4 + 0) * LDQ + d] = (__bf16)v4.x;
            kvt[(seg + c * 4 + 1) * LDQ + d] = (__bf16)v4.y;
            kvt[(seg + c * 4 + 2) * LDQ + d] = (__bf16)v4.z;
            kvt[(seg + c * 4 + 3) * LDQ + d] = (__bf16)v4.w;
        }
    }
    async_wait_all();
    __syncthreads();

    if (t < 128) {  // z per row, from LDS
        float dot = 0.f;
        #pragma unroll
        for (int d = 0; d < Dc; d++) dot += (float)zpq[t * LDQ + d] * ksl[d];
        zrow[t] = 1.0f / (dot + 1e-6f);
    }
    __syncthreads();

    {   // scale pq tile in place by z
        float z = zrow[rr];
        #pragma unroll
        for (int c = 0; c < 4; c++) {
            v8bf vv = *(v8bf*)&zpq[rr * LDQ + d0 + c * 8];
            #pragma unroll
            for (int e = 0; e < 8; e++) vv[e] = (__bf16)(z * (float)vv[e]);
            *(v8bf*)&zpq[rr * LDQ + d0 + c * 8] = vv;
        }
    }
    __syncthreads();

    const int srow = wave * 16;
    v8f acc[4];
    #pragma unroll
    for (int j = 0; j < 4; j++) acc[j] = {};

    #pragma unroll
    for (int ksb = 0; ksb < Dc; ksb += 32) {
        int row = srow + (lane & 15);
        int kk  = ksb + (lane >> 4) * 16;
        v16bf fa = pack2(*(const v8bf*)&zpq[row * LDQ + kk],
                         *(const v8bf*)&zpq[row * LDQ + kk + 8]);
        int kk16 = ksb + (lane >> 4) * 16;
        #pragma unroll
        for (int j = 0; j < 4; j++) {
            int m = j * 16 + (lane & 15);
            v16bf fbv = pack2(*(const v8bf*)&kvt[m * LDQ + kk16],
                              *(const v8bf*)&kvt[m * LDQ + kk16 + 8]);
            acc[j] = __builtin_amdgcn_wmma_f32_16x16x32_bf16(
                false, fa, false, fbv, (short)0, acc[j], false, false);
        }
    }

    #pragma unroll
    for (int j = 0; j < 4; j++) {
        int col = h * Dc + j * 16 + (lane & 15);
        #pragma unroll
        for (int rr2 = 0; rr2 < 8; rr2++) {
            int token = b * Sc + s0 + srow + rr2 + ((lane >> 4) * 8);
            OUT[(size_t)token * Ec + col] = acc[j][rr2];
        }
    }
}

// ---------------------------------------------------------------- launch
extern "C" void kernel_launch(void* const* d_in, const int* in_sizes, int n_in,
                              void* d_out, int out_size, void* d_ws, size_t ws_size,
                              hipStream_t stream) {
    const float* x     = (const float*)d_in[0];
    const float* W_qkv = (const float*)d_in[1];
    const float* b_qkv = (const float*)d_in[2];
    // d_in[3] = W_proj, d_in[4] = b_proj: unused — reference returns attn
    float* out = (float*)d_out;

    __bf16* xh  = (__bf16*)d_ws;
    __bf16* wh  = xh  + (size_t)TOKc * Ec;
    __bf16* pqh = wh  + (size_t)Ec * E3c;
    __bf16* pkh = pqh + (size_t)TOKc * Ec;
    __bf16* vh  = pkh + (size_t)TOKc * Ec;
    float*  kvp = (float*)(vh + (size_t)TOKc * Ec);
    float*  ksp = kvp + (size_t)SPLIT * Bc * Hc * Dc * Dc;
    float*  kvf = ksp + (size_t)SPLIT * Bc * Hc * Dc;
    float*  ksum = kvf + (size_t)Bc * Hc * Dc * Dc;

    f32_to_bf16_v8<<<2048, 256, 0, stream>>>(x, xh, TOKc * Ec / 8);
    f32_to_bf16_v8<<<512, 256, 0, stream>>>(W_qkv, wh, Ec * E3c / 8);
    qkv_gemm_kernel<<<dim3(E3c / 128, TOKc / 128), 256, 0, stream>>>(
        xh, wh, b_qkv, pqh, pkh, vh);
    kv_ksum_kernel<<<Bc * Hc * SPLIT, 256, 0, stream>>>(pkh, vh, kvp, ksp);
    kv_reduce_kernel<<<1024, 256, 0, stream>>>(
        kvp, ksp, kvf, ksum, Bc * Hc * Dc * Dc, Bc * Hc * Dc);
    attn_kernel<<<Bc * Hc * (Sc / 128), 256, 0, stream>>>(pqh, kvf, ksum, out);
}
